// Session2Embedding_20779051778105
// MI455X (gfx1250) — compile-verified, matrix operand order checked by reading, not run on hardware
//
#include <hip/hip_runtime.h>

typedef __attribute__((ext_vector_type(16))) __bf16 v16bf;
typedef __attribute__((ext_vector_type(8)))  __bf16 v8bf;
typedef __attribute__((ext_vector_type(8)))  float  v8f;
typedef __attribute__((ext_vector_type(4)))  unsigned int v4u;
typedef __attribute__((ext_vector_type(8)))  int v8i;
typedef __attribute__((ext_vector_type(4)))  int v4i;

static constexpr int kH = 128;
static constexpr int kPitch = 136;   // padded LDS row pitch in bf16 elements

// ---------------------------------------------------------------------------
// Zero a float buffer (s_g must be re-zeroed every launch: atomics accumulate)
// ---------------------------------------------------------------------------
__global__ void k_zero(float* __restrict__ p, int n) {
    int i = blockIdx.x * blockDim.x + threadIdx.x;
    if (i < n) p[i] = 0.0f;
}

// ---------------------------------------------------------------------------
// Convert W2 (fp32 row-major [H][H]) to bf16 once; consumed via TDM by k_main.
// ---------------------------------------------------------------------------
__global__ void k_cvt_w2(const float* __restrict__ W2, __bf16* __restrict__ W2bf, int n) {
    int i = blockIdx.x * blockDim.x + threadIdx.x;
    if (i < n) W2bf[i] = (__bf16)W2[i];
}

// ---------------------------------------------------------------------------
// last_idx[b] = last position i with batch[i]==b (batch sorted, all nonempty)
// ---------------------------------------------------------------------------
__global__ void k_lastidx(const int* __restrict__ batch, int* __restrict__ last_idx, int n) {
    int i = blockIdx.x * blockDim.x + threadIdx.x;
    if (i >= n) return;
    int b = batch[i];
    if (i == n - 1 || batch[i + 1] != b) last_idx[b] = i;
}

// ---------------------------------------------------------------------------
// Per-session precompute: v_n[b] = x[last_idx[b]],
//                         a1[b]  = W1 @ v_n[b] + b1 + b2   (folds both biases)
// Replaces the 34 GFLOP v_n_rep @ W1^T GEMM with a 134 MFLOP per-segment one.
// ---------------------------------------------------------------------------
__global__ __launch_bounds__(128) void k_prep(
    const float* __restrict__ x, const int* __restrict__ last_idx,
    const float* __restrict__ W1, const float* __restrict__ b1,
    const float* __restrict__ b2,
    float* __restrict__ v_n, float* __restrict__ a1g)
{
    __shared__ float sv[kH];
    const int b = blockIdx.x;
    const int j = threadIdx.x;
    const long li = last_idx[b];
    float v = x[li * kH + j];
    sv[j] = v;
    v_n[(long)b * kH + j] = v;
    __syncthreads();
    const float* w = W1 + (long)j * kH;
    float s = b1[j] + b2[j];
#pragma unroll 8
    for (int k = 0; k < kH; ++k) s += w[k] * sv[k];
    a1g[(long)b * kH + j] = s;
}

// ---------------------------------------------------------------------------
// Fused main pass (one wave = one 16-row tile of x):
//   TDM stages W2(bf16, 32KB) global->LDS once per block, bank-padded by the
//   descriptor's pad fields (4 dwords after every 64 -> pitch 136 halfwords).
//   pre   = x_tile @ W2^T          (bf16 WMMA, f32 accum; B frags from LDS)
//   pre  += a1[batch[row]];  g = sigmoid(pre)
//   alpha = g . Wq + bq            (in-register + lane shuffles; gate never
//                                   touches memory)
//   s_g[batch[row]] += alpha * x[row]   (register runs + f32 atomics into
//                                        the L2-resident 2MB s_g)
// All hot addresses use 32-bit offsets from uniform bases -> GVS scale_offset
// addressing. A sched_barrier keeps all DS/VMEM prefetches ahead of the WMMA
// chain so waits stagger instead of fully serializing.
// ---------------------------------------------------------------------------
__global__ __launch_bounds__(256) void k_main(
    const float* __restrict__ x, const int* __restrict__ batch,
    const __bf16* __restrict__ W2bf, const float* __restrict__ a1g,
    const float* __restrict__ Wq, const float* __restrict__ bq,
    float* __restrict__ s_g)
{
    __shared__ __align__(16) __bf16 shW2[kH * kPitch];   // 34,816 B
    __shared__ int sh_seg[8 * 16];

    const int lane = threadIdx.x & 31;
    const int wave = threadIdx.x >> 5;
    const int tile = blockIdx.x * 8 + wave;
    const int R0 = tile * 16;           // first row of this wave's tile
    const int m  = lane & 15;
    const int hi = lane >> 4;

    if (lane < 16) sh_seg[wave * 16 + lane] = batch[R0 + lane];

    // ---- TDM: wave 0 DMA-loads the 128x128 bf16 W2 tile into LDS with
    // hardware row padding (D# per ISA 8.3/8.4).
    if (wave == 0) {
        const unsigned lds_off = (unsigned)(unsigned long long)(&shW2[0]);
        const unsigned long long ga = (unsigned long long)W2bf;
        v4u g0 = { 1u,                                   // count=1, user D#
                   lds_off,                              // lds_addr (bytes)
                   (unsigned)(ga & 0xffffffffu),         // global_addr[31:0]
                   (unsigned)((ga >> 32) & 0x01ffffffu)  // global_addr[56:32]
                       | 0x80000000u };                  // type=2 ("image")
        // dw0: data_size=1 (2B) <<16 | pad_enable<<20 | pad_interval=5(64dw)<<22
        //      | pad_amount=3(4dw)<<25
        v8i g1 = { (int)0x07510000,
                   (int)(128u << 16),   // tensor_dim0[15:0] in [63:48]
                   (int)(128u << 16),   // tensor_dim0 hi=0 | tensor_dim1[15:0]
                   (int)(128u << 16),   // tensor_dim1 hi=0 | tile_dim0=128
                   (int)128,            // tile_dim1=128, tile_dim2=0
                   (int)128,            // tensor_dim0_stride[31:0]=128
                   0, 0 };              // stride0 hi / tensor_dim1_stride = 0
        v4i gz4 = {0, 0, 0, 0};
        v8i gz8 = {0, 0, 0, 0, 0, 0, 0, 0};
        __builtin_amdgcn_tensor_load_to_lds(g0, g1, gz4, gz4, gz8, 0);
        __builtin_amdgcn_s_wait_tensorcnt(0);
    }
    __syncthreads();

    // ---- A fragments: 16x128 tile of x, fp32 -> bf16 in-register.
    // 16-bit A layout (ISA 7.12.2): lane(hi,m) holds K = 32t+8hi+e (e=0..7)
    // and K = 32t+16+8hi+e in elements 8..15.
    const int rowOff = (R0 + m) * kH;                  // < 2^27, fits int
    v16bf afrag[4];
#pragma unroll
    for (int t = 0; t < 4; ++t) {
        const int p = rowOff + 32 * t + 8 * hi;
#pragma unroll
        for (int e = 0; e < 8; ++e) {
            afrag[t][e]     = (__bf16)x[p + e];
            afrag[t][e + 8] = (__bf16)x[p + 16 + e];
        }
    }

    // segment id per accumulator row (row = r + 8*hi in tile)
    int segr[8];
#pragma unroll
    for (int r = 0; r < 8; ++r) segr[r] = sh_seg[wave * 16 + r + 8 * hi] * kH;

    float palpha[8] = {0.f, 0.f, 0.f, 0.f, 0.f, 0.f, 0.f, 0.f};

    // ---- 8 column tiles of W2^T; B fragment mirrors the A layout, sourced
    // from LDS rows of W2 (row j supplies column j of W2^T).
    for (int nt = 0; nt < 8; ++nt) {
        const int j = nt * 16 + m;                   // output feature column
        const __bf16* rowB = shW2 + j * kPitch;

        // prefetch all 4 B fragments (8 ds_load_b128, in-order returns ->
        // staggered dscnt waits) ...
        v16bf bfrag[4];
#pragma unroll
        for (int t = 0; t < 4; ++t) {
            v8bf blo = *(const v8bf*)(rowB + 32 * t + 8 * hi);        // ds_load_b128
            v8bf bhi = *(const v8bf*)(rowB + 32 * t + 16 + 8 * hi);   // ds_load_b128
            bfrag[t] = __builtin_shufflevector(
                blo, bhi, 0, 1, 2, 3, 4, 5, 6, 7, 8, 9, 10, 11, 12, 13, 14, 15);
        }
        // ... and the a1/Wq gathers, whose VMEM latency hides under the WMMAs
        const float wqj = Wq[j];
        float a1v[8];
#pragma unroll
        for (int r = 0; r < 8; ++r) a1v[r] = a1g[segr[r] + j];

        // keep all prefetches above the WMMA chain
        __builtin_amdgcn_sched_barrier(0);

        v8f acc = {0.f, 0.f, 0.f, 0.f, 0.f, 0.f, 0.f, 0.f};
#pragma unroll
        for (int t = 0; t < 4; ++t)
            acc = __builtin_amdgcn_wmma_f32_16x16x32_bf16(
                false, afrag[t], false, bfrag[t], (short)0, acc, false, false);

        // D layout: acc[r] = out[row r+8hi][col j]
#pragma unroll
        for (int r = 0; r < 8; ++r) {
            float pre = acc[r] + a1v[r];
            float g = __builtin_amdgcn_rcpf(1.0f + __expf(-pre));   // sigmoid
            palpha[r] += g * wqj;
        }
    }

    // reduce alpha partials across the 16 lanes of each half (cols 0..15)
#pragma unroll
    for (int r = 0; r < 8; ++r) {
#pragma unroll
        for (int msk = 1; msk < 16; msk <<= 1)
            palpha[r] += __shfl_xor(palpha[r], msk, 32);
    }
    const float bqv = bq[0];

    // ---- segment-sum of alpha*x. Lane owns 4 contiguous columns; batch is
    // sorted so rows in a tile form at most a few segment runs -> accumulate
    // in registers, flush with global f32 atomics (s_g is 2MB, L2-resident).
    // The x re-read hits L2: this wave touched the same 8KB moments ago.
    const int c = lane * 4;
    float s0 = 0.f, s1 = 0.f, s2 = 0.f, s3 = 0.f;
    int cur = sh_seg[wave * 16];
#pragma unroll
    for (int mm = 0; mm < 16; ++mm) {
        const int sg = sh_seg[wave * 16 + mm];                 // wave-uniform
        const float am = __shfl(palpha[mm & 7], (mm >> 3) * 16, 32) + bqv;
        if (sg != cur) {
            const int o = cur * kH + c;
            atomicAdd(&s_g[o + 0], s0);
            atomicAdd(&s_g[o + 1], s1);
            atomicAdd(&s_g[o + 2], s2);
            atomicAdd(&s_g[o + 3], s3);
            s0 = s1 = s2 = s3 = 0.f;
            cur = sg;
        }
        const float4 xv = *(const float4*)(x + (R0 + mm) * kH + c);
        s0 += am * xv.x; s1 += am * xv.y; s2 += am * xv.z; s3 += am * xv.w;
    }
    const int o = cur * kH + c;
    atomicAdd(&s_g[o + 0], s0);
    atomicAdd(&s_g[o + 1], s1);
    atomicAdd(&s_g[o + 2], s2);
    atomicAdd(&s_g[o + 3], s3);
}

// ---------------------------------------------------------------------------
// s_h[b] = W3 @ [v_n[b] ; s_g[b]] + b3.  One block per session, 128 threads.
// ---------------------------------------------------------------------------
__global__ __launch_bounds__(128) void k_final(
    const float* __restrict__ v_n, const float* __restrict__ s_g,
    const float* __restrict__ W3, const float* __restrict__ b3,
    float* __restrict__ out)
{
    __shared__ float sv[2 * kH];
    const int b = blockIdx.x;
    const int j = threadIdx.x;
    sv[j]       = v_n[(long)b * kH + j];
    sv[kH + j]  = s_g[(long)b * kH + j];
    __syncthreads();
    const float* w = W3 + (long)j * (2 * kH);
    float s = b3[j];
#pragma unroll 8
    for (int k = 0; k < 2 * kH; ++k) s += w[k] * sv[k];
    out[(long)b * kH + j] = s;
}

// ---------------------------------------------------------------------------
extern "C" void kernel_launch(void* const* d_in, const int* in_sizes, int n_in,
                              void* d_out, int out_size, void* d_ws, size_t ws_size,
                              hipStream_t stream) {
    const float* x     = (const float*)d_in[0];
    const int*   batch = (const int*)  d_in[1];
    const float* W1    = (const float*)d_in[2];
    const float* b1    = (const float*)d_in[3];
    const float* W2    = (const float*)d_in[4];
    const float* b2    = (const float*)d_in[5];
    const float* Wq    = (const float*)d_in[6];
    const float* bq    = (const float*)d_in[7];
    const float* W3    = (const float*)d_in[8];
    const float* b3    = (const float*)d_in[9];
    float* out = (float*)d_out;

    const int N = in_sizes[1];          // number of nodes
    const int B = out_size / kH;        // number of sessions

    // workspace layout (256B-aligned slabs)
    char* ws = (char*)d_ws;
    size_t off = 0;
    auto alloc = [&](size_t bytes) {
        char* p = ws + off;
        off = (off + bytes + 255) & ~(size_t)255;
        return p;
    };
    int*    last_idx = (int*)   alloc((size_t)B * sizeof(int));
    float*  v_n      = (float*) alloc((size_t)B * kH * sizeof(float));
    float*  a1g      = (float*) alloc((size_t)B * kH * sizeof(float));
    float*  s_g      = (float*) alloc((size_t)B * kH * sizeof(float));
    __bf16* W2bf     = (__bf16*)alloc((size_t)kH * kH * sizeof(__bf16));

    const int BH = B * kH;
    k_zero<<<(BH + 255) / 256, 256, 0, stream>>>(s_g, BH);
    k_cvt_w2<<<(kH * kH + 255) / 256, 256, 0, stream>>>(W2, W2bf, kH * kH);
    k_lastidx<<<(N + 255) / 256, 256, 0, stream>>>(batch, last_idx, N);
    k_prep<<<B, 128, 0, stream>>>(x, last_idx, W1, b1, b2, v_n, a1g);

    // one wave per 16-row tile, 8 waves per block; N = 1M is divisible by 128
    const int nBlocks = N / 128;
    k_main<<<nBlocks, 256, 0, stream>>>(x, batch, W2bf, a1g, Wq, bq, s_g);

    k_final<<<B, 128, 0, stream>>>(v_n, s_g, W3, b3, out);
}
